// CompositionalMlp_32263794327738
// MI455X (gfx1250) — matrix operand order repaired and checked by reading, not compiled, not run on hardware
//
#include <hip/hip_runtime.h>

typedef __attribute__((ext_vector_type(8)))  _Float16 h8;
typedef __attribute__((ext_vector_type(16))) _Float16 v16h;
typedef __attribute__((ext_vector_type(8)))  float    v8f;

#define DEVINL __device__ __forceinline__

static constexpr int NEXP = 4;
static constexpr int F    = 32;
static constexpr int H    = 256;
static constexpr int DMID = 128;
static constexpr int DOUT = 8;
static constexpr int INW  = 144;   // 4*32 feats + 4*4 one-hots

static constexpr int ROWS = 128;   // rows per block
static constexpr int MT   = ROWS / 16;

// LDS strides in halves (all multiples of 8 -> every b128 access 16B aligned)
static constexpr int SXJ = 40;     // node features [128 x 32]
static constexpr int SXP = 136;    // node outputs   [128 x 128] (x2, ping-pong)
static constexpr int SH  = 264;    // hidden buffers [128 x 256]
static constexpr int NXJ = ROWS * SXJ;
static constexpr int NXP = ROWS * SXP;
static constexpr int NH  = ROWS * SH;
static constexpr size_t SMEM_BYTES =
    size_t(NXJ + 2 * NXP + 2 * NH) * 2 + size_t(ROWS) * 16 * 4;  // ~218 KB

DEVINL v8f vzero() { v8f z = {0.f,0.f,0.f,0.f,0.f,0.f,0.f,0.f}; return z; }

DEVINL v8f wmma16(v16h a, v16h b, v8f c) {
  return __builtin_amdgcn_wmma_f32_16x16x32_f16(false, a, false, b, (short)0, c,
                                                false, false);
}

// A operand (16-bit 16x32): 16 contiguous K halves per lane
DEVINL v16h ldA16(const _Float16* p) {
  h8 lo = *(const h8*)(p);
  h8 hi = *(const h8*)(p + 8);
  return __builtin_shufflevector(lo, hi, 0,1,2,3,4,5,6,7,8,9,10,11,12,13,14,15);
}
// B operand (16-bit 32x16, column-major feed): 16 contiguous K halves per lane
DEVINL v16h ldB16(const _Float16* p) {
  h8 lo = *(const h8*)(p);
  h8 hi = *(const h8*)(p + 8);
  return __builtin_shufflevector(lo, hi, 0,1,2,3,4,5,6,7,8,9,10,11,12,13,14,15);
}

// dst[ROWS x Nout] = relu(concat(a0,a1) @ Wt^T + bias), f16 into LDS.
// wt: [Nout][Ktot] f16 (K contiguous). Wave w owns N-tiles [w*NTPW, w*NTPW+NTPW).
template<int K0C, int K1C, int NTPW>
DEVINL void layer_to_lds(const _Float16* a0, int s0,
                         const _Float16* a1, int s1,
                         const _Float16* wt, int Ktot,
                         const float* bias,
                         _Float16* dst, int ds,
                         int lane, int wave)
{
  const int rlo  = lane & 15;
  const int khiA = (lane >> 4) << 4;
  const int kbB  = (lane >> 4) << 4;
  const int rhi  = (lane >> 4) << 3;
#pragma unroll 1
  for (int i = 0; i < NTPW; ++i) {
    const int nt = wave * NTPW + i;
    const _Float16* wLane = wt + (nt * 16 + rlo) * Ktot + kbB;
    const float bv = bias[nt * 16 + rlo];
    v8f acc[MT];
#pragma unroll
    for (int mt = 0; mt < MT; ++mt) acc[mt] = vzero();
#pragma unroll 1
    for (int c = 0; c < K0C; ++c) {
      v16h bm = ldB16(wLane + c * 32);
      const _Float16* aLane = a0 + rlo * s0 + c * 32 + khiA;
#pragma unroll
      for (int mt = 0; mt < MT; ++mt) {
        v16h am = ldA16(aLane + mt * 16 * s0);
        acc[mt] = wmma16(am, bm, acc[mt]);
      }
    }
#pragma unroll 1
    for (int c = 0; c < K1C; ++c) {
      v16h bm = ldB16(wLane + (K0C + c) * 32);
      const _Float16* aLane = a1 + rlo * s1 + c * 32 + khiA;
#pragma unroll
      for (int mt = 0; mt < MT; ++mt) {
        v16h am = ldA16(aLane + mt * 16 * s1);
        acc[mt] = wmma16(am, bm, acc[mt]);
      }
    }
    const int col = nt * 16 + rlo;
#pragma unroll
    for (int mt = 0; mt < MT; ++mt) {          // fully unrolled: keep acc[] in VGPRs
      const int rbase = mt * 16 + rhi;
#pragma unroll
      for (int v = 0; v < 8; ++v) {
        float x = acc[mt][v] + bv;
        x = fmaxf(x, 0.0f);
        dst[(rbase + v) * ds + col] = (_Float16)x;
      }
    }
  }
}

// Node-final layer (Nout=128): expert-selected relu(h @ W1 + b), accumulated
// into the f16 LDS output buffer xnew (one-hot -> exactly one non-zero term).
template<int KC>
DEVINL void final_layer_accum(const _Float16* hbuf, int sh,
                              const _Float16* wt, int Ktot,
                              const float* bias,
                              const float* ohL, int ohc, bool first,
                              _Float16* xnew, int lane, int wave)
{
  const int rlo  = lane & 15;
  const int khiA = (lane >> 4) << 4;
  const int kbB  = (lane >> 4) << 4;
  const int rhi  = (lane >> 4) << 3;
  const int nt   = wave;   // 8 N-tiles, one per wave
  const _Float16* wLane = wt + (nt * 16 + rlo) * Ktot + kbB;
  const float bv = bias[nt * 16 + rlo];
  v8f acc[MT];
#pragma unroll
  for (int mt = 0; mt < MT; ++mt) acc[mt] = vzero();
#pragma unroll 1
  for (int c = 0; c < KC; ++c) {
    v16h bm = ldB16(wLane + c * 32);
    const _Float16* aLane = hbuf + rlo * sh + c * 32 + khiA;
#pragma unroll
    for (int mt = 0; mt < MT; ++mt) {
      v16h am = ldA16(aLane + mt * 16 * sh);
      acc[mt] = wmma16(am, bm, acc[mt]);
    }
  }
  const int colX = nt * 16 + rlo;
#pragma unroll
  for (int mt = 0; mt < MT; ++mt) {            // fully unrolled: acc stays in VGPRs
    const int rbase = mt * 16 + rhi;
#pragma unroll
    for (int v = 0; v < 8; ++v) {
      float x = fmaxf(acc[mt][v] + bv, 0.0f);
      float s = ohL[(rbase + v) * 16 + ohc];
      const int idx = (rbase + v) * SXP + colX;
      float prev = first ? 0.0f : (float)xnew[idx];
      xnew[idx] = (_Float16)(prev + s * x);
    }
  }
}

// Node-3 head (Nout padded to 16, no relu). Wave w owns M-tile w, N-tile 0.
DEVINL void final_node3(const _Float16* hbuf, int sh,
                        const _Float16* wt, const float* biasP,
                        const float* ohL, int ohc,
                        v8f& accO, int lane, int wave)
{
  const int rlo  = lane & 15;
  const int khiA = (lane >> 4) << 4;
  const int kbB  = (lane >> 4) << 4;
  const _Float16* wLane = wt + rlo * H + kbB;
  const float bv = biasP[rlo];
  v8f acc = vzero();
#pragma unroll 1
  for (int c = 0; c < 8; ++c) {
    v16h bm = ldB16(wLane + c * 32);
    v16h am = ldA16(hbuf + (wave * 16 + rlo) * sh + c * 32 + khiA);
    acc = wmma16(am, bm, acc);
  }
  const int rbase = wave * 16 + ((lane >> 4) << 3);
#pragma unroll
  for (int v = 0; v < 8; ++v) {
    float x = acc[v] + bv;
    accO[v] += ohL[(rbase + v) * 16 + ohc] * x;
  }
}

DEVINL void load_xj(const float* input, int row0, int node, _Float16* xj, int tid) {
#pragma unroll 1
  for (int i = 0; i < (ROWS * F) / 256; ++i) {
    int idx = tid + i * 256;
    int r = idx >> 5, c = idx & 31;
    xj[r * SXJ + c] = (_Float16)input[(size_t)(row0 + r) * INW + node * F + c];
  }
}

DEVINL void load_oh(const float* input, int row0, float* ohL, int tid) {
#pragma unroll 1
  for (int i = 0; i < (ROWS * 16) / 256; ++i) {
    int idx = tid + i * 256;
    int r = idx >> 4, c = idx & 15;
    ohL[idx] = input[(size_t)(row0 + r) * INW + 4 * F + c];
  }
}

DEVINL void middle_node(const float* input, int row0, int node,
                        const _Float16* WtP, const float* bP,
                        const _Float16* Wt0, const float* b0,
                        const _Float16* Wt1, const float* b1,
                        _Float16* xj, const _Float16* xprev, _Float16* xnew,
                        _Float16* hpre, _Float16* h2,
                        const float* ohL, int tid, int lane, int wave)
{
  load_xj(input, row0, node, xj, tid);
  __syncthreads();
#pragma unroll 1
  for (int e = 0; e < NEXP; ++e) {
    layer_to_lds<1, 0, 2>(xj, SXJ, nullptr, 0, WtP + e * H * F, F, bP + e * H,
                          hpre, SH, lane, wave);
    __syncthreads();
    layer_to_lds<4, 8, 2>(xprev, SXP, hpre, SH, Wt0 + e * H * (DMID + H),
                          DMID + H, b0 + e * H, h2, SH, lane, wave);
    __syncthreads();
    final_layer_accum<8>(h2, SH, Wt1 + e * DMID * H, H, b1 + e * DMID,
                         ohL, node * 4 + e, e == 0, xnew, lane, wave);
    __syncthreads();
  }
}

__global__ __launch_bounds__(256, 1) void moe_fwd(
    const float* __restrict__ input, float* __restrict__ out,
    const _Float16* Wt00, const float* b00, const _Float16* Wt01, const float* b01,
    const _Float16* WtP1, const float* bP1, const _Float16* Wt10, const float* b10,
    const _Float16* Wt11, const float* b11,
    const _Float16* WtP2, const float* bP2, const _Float16* Wt20, const float* b20,
    const _Float16* Wt21, const float* b21,
    const _Float16* WtP3, const float* bP3, const _Float16* Wt30, const float* b30,
    const _Float16* Wt31, const float* b31p)
{
  extern __shared__ char smem[];
  _Float16* xj    = (_Float16*)smem;
  _Float16* xA    = xj + NXJ;     // node-output ping buffer [128 x 128]
  _Float16* xB    = xA + NXP;     // node-output pong buffer [128 x 128]
  _Float16* hpre  = xB + NXP;
  _Float16* h2    = hpre + NH;
  float*    ohL   = (float*)(h2 + NH);

  const int tid  = threadIdx.x;
  const int lane = tid & 31;
  const int wave = tid >> 5;
  const int row0 = blockIdx.x * ROWS;

  load_oh(input, row0, ohL, tid);
  load_xj(input, row0, 0, xj, tid);
  __syncthreads();

  // node 0: relu(x0 @ W0_0 + b) -> relu(h @ W0_1 + b), expert-select -> xA
#pragma unroll 1
  for (int e = 0; e < NEXP; ++e) {
    layer_to_lds<1, 0, 2>(xj, SXJ, nullptr, 0, Wt00 + e * H * F, F, b00 + e * H,
                          hpre, SH, lane, wave);
    __syncthreads();
    final_layer_accum<8>(hpre, SH, Wt01 + e * DMID * H, H, b01 + e * DMID,
                         ohL, e, e == 0, xA, lane, wave);
    __syncthreads();
  }

  // node 1: reads xA, writes xB; node 2: reads xB, writes xA
  middle_node(input, row0, 1, WtP1, bP1, Wt10, b10, Wt11, b11,
              xj, xA, xB, hpre, h2, ohL, tid, lane, wave);
  middle_node(input, row0, 2, WtP2, bP2, Wt20, b20, Wt21, b21,
              xj, xB, xA, hpre, h2, ohL, tid, lane, wave);

  // node 3 (output head, no final relu): reads xA
  load_xj(input, row0, 3, xj, tid);
  __syncthreads();
  v8f accO = vzero();
#pragma unroll 1
  for (int e = 0; e < NEXP; ++e) {
    layer_to_lds<1, 0, 2>(xj, SXJ, nullptr, 0, WtP3 + e * H * F, F, bP3 + e * H,
                          hpre, SH, lane, wave);
    __syncthreads();
    layer_to_lds<4, 8, 2>(xA, SXP, hpre, SH, Wt30 + e * H * (DMID + H),
                          DMID + H, b30 + e * H, h2, SH, lane, wave);
    __syncthreads();
    final_node3(h2, SH, Wt31 + e * 16 * H, b31p + e * 16, ohL, 12 + e,
                accO, lane, wave);
    __syncthreads();
  }
  const int col = lane & 15;
  if (col < DOUT) {
    const int rbase = wave * 16 + ((lane >> 4) << 3);
#pragma unroll
    for (int v = 0; v < 8; ++v)
      out[(size_t)(row0 + rbase + v) * DOUT + col] = accO[v];
  }
}

// ---- weight prep: f32 [E][K][N] -> f16 [E][N][K] (K contiguous per column) ----
__global__ void prep_transpose(const float* __restrict__ src, _Float16* __restrict__ dst,
                               int K, int N, int total) {
  int idx = blockIdx.x * blockDim.x + threadIdx.x;
  if (idx >= total) return;
  int n = idx % N;
  int k = (idx / N) % K;
  int e = idx / (N * K);
  dst[((size_t)e * N + n) * K + k] = (_Float16)src[idx];
}

__global__ void prep_w31(const float* __restrict__ src, _Float16* __restrict__ dst) {
  int idx = blockIdx.x * blockDim.x + threadIdx.x;  // dst [4][16][256]
  if (idx >= 4 * 16 * 256) return;
  int k = idx & 255;
  int n = (idx >> 8) & 15;
  int e = idx >> 12;
  float v = (n < DOUT) ? src[((size_t)e * 256 + k) * DOUT + n] : 0.0f;
  dst[idx] = (_Float16)v;
}

__global__ void prep_b31(const float* __restrict__ src, float* __restrict__ dst) {
  int idx = threadIdx.x;
  if (idx >= 64) return;
  int e = idx >> 4, n = idx & 15;
  dst[idx] = (n < DOUT) ? src[e * DOUT + n] : 0.0f;
}

extern "C" void kernel_launch(void* const* d_in, const int* in_sizes, int n_in,
                              void* d_out, int out_size, void* d_ws, size_t ws_size,
                              hipStream_t stream) {
  (void)in_sizes; (void)n_in; (void)out_size; (void)ws_size;
  const float* input = (const float*)d_in[0];

  _Float16* ws = (_Float16*)d_ws;
  size_t off = 0;
  auto take = [&](size_t n) { _Float16* p = ws + off; off += n; return p; };
  _Float16* Wt00 = take((size_t)4 * 256 * 32);
  _Float16* Wt01 = take((size_t)4 * 128 * 256);
  _Float16* WtP1 = take((size_t)4 * 256 * 32);
  _Float16* Wt10 = take((size_t)4 * 256 * 384);
  _Float16* Wt11 = take((size_t)4 * 128 * 256);
  _Float16* WtP2 = take((size_t)4 * 256 * 32);
  _Float16* Wt20 = take((size_t)4 * 256 * 384);
  _Float16* Wt21 = take((size_t)4 * 128 * 256);
  _Float16* WtP3 = take((size_t)4 * 256 * 32);
  _Float16* Wt30 = take((size_t)4 * 256 * 384);
  _Float16* Wt31 = take((size_t)4 * 16 * 256);
  float* b31p = (float*)(ws + off);
  off += 128;

  auto tp = [&](const void* src, _Float16* dst, int K, int N) {
    int total = 4 * K * N;
    prep_transpose<<<(total + 255) / 256, 256, 0, stream>>>((const float*)src, dst,
                                                            K, N, total);
  };
  tp(d_in[1],  Wt00, 32, 256);
  tp(d_in[3],  Wt01, 256, 128);
  tp(d_in[5],  WtP1, 32, 256);
  tp(d_in[7],  Wt10, 384, 256);
  tp(d_in[9],  Wt11, 256, 128);
  tp(d_in[11], WtP2, 32, 256);
  tp(d_in[13], Wt20, 384, 256);
  tp(d_in[15], Wt21, 256, 128);
  tp(d_in[17], WtP3, 32, 256);
  tp(d_in[19], Wt30, 384, 256);
  prep_w31<<<(4 * 16 * 256 + 255) / 256, 256, 0, stream>>>((const float*)d_in[21], Wt31);
  prep_b31<<<1, 64, 0, stream>>>((const float*)d_in[22], b31p);

  static_assert(SMEM_BYTES < 320u * 1024u, "LDS budget");
  hipFuncSetAttribute(reinterpret_cast<const void*>(moe_fwd),
                      hipFuncAttributeMaxDynamicSharedMemorySize, (int)SMEM_BYTES);
  moe_fwd<<<65536 / ROWS, 256, SMEM_BYTES, stream>>>(
      input, (float*)d_out,
      Wt00, (const float*)d_in[2],  Wt01, (const float*)d_in[4],
      WtP1, (const float*)d_in[6],  Wt10, (const float*)d_in[8],
      Wt11, (const float*)d_in[10],
      WtP2, (const float*)d_in[12], Wt20, (const float*)d_in[14],
      Wt21, (const float*)d_in[16],
      WtP3, (const float*)d_in[18], Wt30, (const float*)d_in[20],
      Wt31, b31p);
}
